// GAT_Placement_82978768158888
// MI455X (gfx1250) — compile-verified, matrix operand order checked by reading, not run on hardware
//
#include <hip/hip_runtime.h>
#include <hip/hip_bf16.h>

#define NNODES 50000
#define NEDGES 1600000
#define HDIM   128
#define NU_    49232
#define NM_    512
#define NS_    256

typedef __attribute__((ext_vector_type(16))) _Float16 v16h;
typedef __attribute__((ext_vector_type(8)))  float    v8f;

// ---------------- fills (graph-capture-safe init) ----------------
__global__ void fill_f32(float* p, float v, int n) {
    int i = blockIdx.x * blockDim.x + threadIdx.x;
    if (i < n) p[i] = v;
}
__global__ void fill_u32(unsigned* p, unsigned v, int n) {
    int i = blockIdx.x * blockDim.x + threadIdx.x;
    if (i < n) p[i] = v;
}

// ---------------- WMMA GEMM: C[M,128] = A[M,K] @ B[K,128], row-major f32 ----------
// W^T staged into LDS as f16 (padded rows, bank-conflict-free fragment loads).
// grid.x = M/16, block = 256 (8 waves); wave w owns n-tile w (Nn = 128 = 8 tiles).
template <int K>
__global__ void gemm_rm_lds(const float* __restrict__ A, const float* __restrict__ B,
                            float* __restrict__ C) {
    constexpr int Nn  = 128;
    constexpr int PAD = 8;                       // row stride 136 f16 = 272B (16B-aligned, odd banks)
    __shared__ _Float16 WT[Nn][K + PAD];

    const int tid = threadIdx.x;
    // stage B^T into LDS as f16: coalesced global reads over n for each k
    for (int idx = tid; idx < Nn * K; idx += 256) {
        int k = idx >> 7;       // idx / 128
        int n = idx & 127;      // idx % 128
        WT[n][k] = (_Float16)B[idx];
    }
    __syncthreads();

    const int wave  = tid >> 5;
    const int lane  = tid & 31;
    const int m0    = blockIdx.x * 16;
    const int lrow  = lane & 15;
    const int khalf = (lane >> 4) * 8;           // lanes 16-31: K offset +8 / +24
    const int n0    = wave * 16;

    v8f c = {};
#pragma unroll
    for (int k0 = 0; k0 < K; k0 += 32) {
        v16h a, b;
        const float* arow = A + (size_t)(m0 + lrow) * K + k0 + khalf;
#pragma unroll
        for (int i = 0; i < 8; ++i) {
            a[i]     = (_Float16)arow[i];
            a[8 + i] = (_Float16)arow[16 + i];
        }
        const _Float16* wrow = &WT[n0 + lrow][k0 + khalf];
#pragma unroll
        for (int i = 0; i < 8; ++i) {
            b[i]     = wrow[i];
            b[8 + i] = wrow[16 + i];
        }
        c = __builtin_amdgcn_wmma_f32_16x16x32_f16(false, a, false, b,
                                                   (short)0, c, false, false);
    }
    // D layout: VGPR r -> row m0 + r + (lane>=16 ? 8 : 0), col n0 + (lane&15)
    float* crow = C + (size_t)(m0 + (lane >> 4) * 8) * Nn + n0 + lrow;
#pragma unroll
    for (int r = 0; r < 8; ++r) crow[(size_t)r * Nn] = c[r];
}

// scores = A[M,K] @ S[N,K]^T + bb  (B columns are contiguous rows of S)
template <int K>
__global__ void gemm_abt_bias(const float* __restrict__ A, const float* __restrict__ S,
                              const float* __restrict__ bbp, float* __restrict__ C,
                              int Nn) {
    const int wave  = threadIdx.x >> 5;
    const int lane  = threadIdx.x & 31;
    const int m0    = blockIdx.x * 16;
    const int lrow  = lane & 15;
    const int khalf = (lane >> 4) * 8;
    const int ntile = Nn >> 4;
    const float bb  = bbp[0];
    for (int nt = wave; nt < ntile; nt += 8) {
        const int n0 = nt * 16;
        v8f c = {};
#pragma unroll
        for (int k0 = 0; k0 < K; k0 += 32) {
            v16h a, b;
            const float* arow = A + (size_t)(m0 + lrow) * K + k0 + khalf;
            const float* brow = S + (size_t)(n0 + lrow) * K + k0 + khalf;
#pragma unroll
            for (int i = 0; i < 8; ++i) {
                a[i]     = (_Float16)arow[i];
                a[8 + i] = (_Float16)arow[16 + i];
                b[i]     = (_Float16)brow[i];
                b[8 + i] = (_Float16)brow[16 + i];
            }
            c = __builtin_amdgcn_wmma_f32_16x16x32_f16(false, a, false, b,
                                                       (short)0, c, false, false);
        }
        float* crow = C + (size_t)(m0 + (lane >> 4) * 8) * Nn + n0 + lrow;
#pragma unroll
        for (int r = 0; r < 8; ++r) crow[(size_t)r * Nn] = c[r] + bb;
    }
}

// ---------------- attention logit GEMVs ----------------
__global__ void compute_elr(const float* __restrict__ h, const float* __restrict__ al,
                            const float* __restrict__ ar, float* el, float* er, int n) {
    int i = blockIdx.x * blockDim.x + threadIdx.x;
    if (i >= n) return;
    const float* row = h + (size_t)i * HDIM;
    float sl = 0.f, sr = 0.f;
#pragma unroll 4
    for (int j = 0; j < HDIM; ++j) {
        float v = row[j];
        sl += v * al[j];
        sr += v * ar[j];
    }
    el[i] = sl;
    er[i] = sr;
}

// ---------------- edge softmax ----------------
__device__ __forceinline__ unsigned f32_ord(float f) {
    unsigned b = __float_as_uint(f);
    return (b & 0x80000000u) ? ~b : (b | 0x80000000u);
}
__device__ __forceinline__ float ord_f32(unsigned u) {
    unsigned b = (u & 0x80000000u) ? (u & 0x7FFFFFFFu) : ~u;
    return __uint_as_float(b);
}

__global__ void edge_max(const int* __restrict__ src, const int* __restrict__ dst,
                         const float* __restrict__ el, const float* __restrict__ er,
                         unsigned* __restrict__ menc) {
    int e = blockIdx.x * blockDim.x + threadIdx.x;
    if (e >= NEDGES) return;
    int s_ = src[e], d_ = dst[e];
    float x = el[d_] + er[s_];
    x = (x > 0.f) ? x : 0.2f * x;  // leaky_relu 0.2
    atomicMax(&menc[d_], f32_ord(x));
}

__global__ void edge_exp(const int* __restrict__ src, const int* __restrict__ dst,
                         const float* __restrict__ el, const float* __restrict__ er,
                         const unsigned* __restrict__ menc, float* __restrict__ ssum,
                         float* __restrict__ wE) {
    int e = blockIdx.x * blockDim.x + threadIdx.x;
    if (e >= NEDGES) return;
    int s_ = src[e], d_ = dst[e];
    float x = el[d_] + er[s_];
    x = (x > 0.f) ? x : 0.2f * x;
    float w = __expf(x - ord_f32(menc[d_]));
    wE[e] = w;
    atomicAdd(&ssum[d_], w);
}

// one wave per edge: lane handles 4 contiguous columns (float4 gather + 4 atomic adds)
__global__ void edge_agg(const int* __restrict__ src, const int* __restrict__ dst,
                         const float* __restrict__ h, const float* __restrict__ ssum,
                         const float* __restrict__ wE, float* __restrict__ out) {
    int wid  = (int)((blockIdx.x * (unsigned)blockDim.x + threadIdx.x) >> 5);
    int lane = threadIdx.x & 31;
    if (wid >= NEDGES) return;
    int s_ = src[wid], d_ = dst[wid];
    float attn = wE[wid] / (ssum[d_] + 1e-10f);
    const float4 hv = *(const float4*)(h + (size_t)s_ * HDIM + lane * 4);
    float* o = out + (size_t)d_ * HDIM + lane * 4;
    atomicAdd(o + 0, hv.x * attn);
    atomicAdd(o + 1, hv.y * attn);
    atomicAdd(o + 2, hv.z * attn);
    atomicAdd(o + 3, hv.w * attn);
}

__global__ void bias_act(float* __restrict__ out, const float* __restrict__ b,
                         int total, int relu) {
    int i = blockIdx.x * blockDim.x + threadIdx.x;
    if (i >= total) return;
    float v = out[i] + b[i & (HDIM - 1)];
    if (relu) v = fmaxf(v, 0.f);
    out[i] = v;
}

extern "C" void kernel_launch(void* const* d_in, const int* in_sizes, int n_in,
                              void* d_out, int out_size, void* d_ws, size_t ws_size,
                              hipStream_t stream) {
    (void)in_sizes; (void)n_in; (void)out_size; (void)ws_size;
    const float* x   = (const float*)d_in[0];
    const int*   ei  = (const int*)d_in[1];
    const int*   src = ei;             // edge_index[0]
    const int*   dst = ei + NEDGES;    // edge_index[1]
    const float* W[3]    = {(const float*)d_in[2], (const float*)d_in[6],  (const float*)d_in[10]};
    const float* al[3]   = {(const float*)d_in[3], (const float*)d_in[7],  (const float*)d_in[11]};
    const float* ar[3]   = {(const float*)d_in[4], (const float*)d_in[8],  (const float*)d_in[12]};
    const float* bias[3] = {(const float*)d_in[5], (const float*)d_in[9],  (const float*)d_in[13]};
    const float* Wb = (const float*)d_in[14];
    const float* bb = (const float*)d_in[15];

    float*    hbuf = (float*)d_ws;                         // N*128
    float*    buf0 = hbuf + (size_t)NNODES * HDIM;         // N*128
    float*    buf1 = buf0 + (size_t)NNODES * HDIM;         // N*128
    float*    el   = buf1 + (size_t)NNODES * HDIM;         // N
    float*    er   = el + NNODES;                          // N
    unsigned* menc = (unsigned*)(er + NNODES);             // N
    float*    ssum = (float*)(menc + NNODES);              // N
    float*    wE   = ssum + NNODES;                        // E
    float*    tbuf = wE + NEDGES;                          // 512*128

    const float* in = x;
    float* outs[3] = {buf0, buf1, buf0};
    for (int l = 0; l < 3; ++l) {
        float* out = outs[l];
        gemm_rm_lds<HDIM><<<NNODES / 16, 256, 0, stream>>>(in, W[l], hbuf);
        compute_elr<<<(NNODES + 255) / 256, 256, 0, stream>>>(hbuf, al[l], ar[l], el, er, NNODES);
        fill_u32<<<(NNODES + 255) / 256, 256, 0, stream>>>(menc, 0u, NNODES);
        fill_f32<<<(NNODES + 255) / 256, 256, 0, stream>>>(ssum, 0.f, NNODES);
        fill_f32<<<(NNODES * HDIM + 255) / 256, 256, 0, stream>>>(out, 0.f, NNODES * HDIM);
        edge_max<<<NEDGES / 256, 256, 0, stream>>>(src, dst, el, er, menc);
        edge_exp<<<NEDGES / 256, 256, 0, stream>>>(src, dst, el, er, menc, ssum, wE);
        edge_agg<<<NEDGES / 8, 256, 0, stream>>>(src, dst, hbuf, ssum, wE, out);
        bias_act<<<(NNODES * HDIM + 255) / 256, 256, 0, stream>>>(out, bias[l], NNODES * HDIM, l < 2 ? 1 : 0);
        in = out;
    }

    // final bilinear scoring: scores[512,256] = model @ Wb @ server^T + bb
    const float* model  = buf0 + (size_t)NU_ * HDIM;
    const float* server = buf0 + (size_t)(NU_ + NM_) * HDIM;
    gemm_rm_lds<HDIM><<<NM_ / 16, 256, 0, stream>>>(model, Wb, tbuf);
    gemm_abt_bias<HDIM><<<NM_ / 16, 256, 0, stream>>>(tbuf, server, bb, (float*)d_out, NS_);
}
